// HGNN_78048145703103
// MI455X (gfx1250) — matrix operand order repaired and checked by reading, not compiled, not run on hardware
//
#include <hip/hip_runtime.h>
#include <cstdint>
#include <cstddef>

// ---------------------------------------------------------------------------
// CDNA5 / gfx1250 HGNN pipeline. wave32, WMMA bf16 16x16x32 for all GEMMs,
// double-buffered async global->LDS staging (GLOBAL_LOAD_ASYNC_TO_LDS_B128,
// s_wait_asynccnt pipelined against v_wmma).
// ---------------------------------------------------------------------------

typedef __attribute__((ext_vector_type(16))) __bf16 v16bf;
typedef __attribute__((ext_vector_type(2)))  __bf16 v2bf;
typedef __attribute__((ext_vector_type(8)))  float  v8f;
typedef __attribute__((ext_vector_type(2)))  float  v2f;
typedef __attribute__((ext_vector_type(4)))  int    i32x4;

#define ORD_NEG_INF 0x007FFFFFu   // f32_ord(-inf)

#if __has_builtin(__builtin_amdgcn_global_load_async_to_lds_b128)
#define USE_ASYNC_LDS 1
#endif

#define GLOBAL_AS __attribute__((address_space(1)))
#define LDS_AS    __attribute__((address_space(3)))

__device__ __forceinline__ unsigned f32_ord(float f) {
  unsigned u = __float_as_uint(f);
  return (u & 0x80000000u) ? ~u : (u | 0x80000000u);
}
__device__ __forceinline__ float ord_f32(unsigned u) {
  return (u & 0x80000000u) ? __uint_as_float(u & 0x7fffffffu)
                           : __uint_as_float(~u);
}
// packed f32x2 -> bf16x2 (RNE); lowers to v_cvt_pk_bf16_f32
__device__ __forceinline__ unsigned pkbf(float a, float b) {
  v2f f; f.x = a; f.y = b;
  v2bf h = __builtin_convertvector(f, v2bf);
  union { v2bf h; unsigned u; } c; c.h = h;
  return c.u;
}
__device__ __forceinline__ float leaky02(float x) { return x > 0.f ? x : 0.2f * x; }

__device__ __forceinline__ float waveReduceSum(float v) {
  #pragma unroll
  for (int o = 16; o; o >>= 1) v += __shfl_down(v, o);
  return v;
}
__device__ __forceinline__ float blockReduceSum(float v, float* sh) {
  v = waveReduceSum(v);
  int lane = threadIdx.x & 31, wid = threadIdx.x >> 5;
  if (!lane) sh[wid] = v;
  __syncthreads();
  float r = (threadIdx.x < (blockDim.x >> 5)) ? sh[threadIdx.x] : 0.f;
  if (wid == 0) {
    #pragma unroll
    for (int o = 4; o; o >>= 1) r += __shfl_down(r, o);
    if (!lane) sh[0] = r;
  }
  __syncthreads();
  float out = sh[0];
  __syncthreads();
  return out;
}

#ifdef USE_ASYNC_LDS
__device__ __forceinline__ void async_b128(const float* g, float* l) {
  __builtin_amdgcn_global_load_async_to_lds_b128(
      (GLOBAL_AS i32x4*)g, (LDS_AS i32x4*)l, 0, 0);
}
__device__ __forceinline__ void wait_async_le6() {
#if __has_builtin(__builtin_amdgcn_s_wait_asynccnt)
  __builtin_amdgcn_s_wait_asynccnt(6);
#else
  asm volatile("s_wait_asynccnt 6" ::: "memory");
#endif
}
__device__ __forceinline__ void wait_async_le0() {
#if __has_builtin(__builtin_amdgcn_s_wait_asynccnt)
  __builtin_amdgcn_s_wait_asynccnt(0);
#else
  asm volatile("s_wait_asynccnt 0" ::: "memory");
#endif
}
#endif

// ---------------------------------------------------------------------------
// GEMM: Cout[m, n] = A[m, :K] . W[n, :K] + bias[n]
// Block tile 128x64, 8 waves (4x2), wave tile 32x32, K-step 32.
// f32 tiles double-buffered in LDS via async DMA; bf16 WMMA compute.
// LDS row stride 36 floats: 36*i mod 64 are distinct multiples of 4 ->
// conflict-free b128 access for staging and both fragment patterns.
// Each wave has exactly 6 async loads in flight per tile (4 A + 2 B);
// async loads retire in order, so s_wait_asynccnt<=6 after issuing tile i+1
// guarantees tile i has fully landed in LDS.
// ---------------------------------------------------------------------------
__global__ __launch_bounds__(256) void k_gemm_bf16(
    const float* __restrict__ A, const float* __restrict__ W,
    const float* __restrict__ bias, float* __restrict__ Cout,
    int M, int Nn, int K, int ldc)
{
  const int tid  = threadIdx.x;
  const int lane = tid & 31;
  const int wid  = tid >> 5;
  const int wm   = wid & 3;       // wave row group 0..3
  const int wn   = wid >> 2;      // wave col group 0..1
  const int lrow = lane & 15;
  const int lhi  = lane >> 4;     // half-wave select
  const int blockRow = blockIdx.x * 128;
  const int blockCol = blockIdx.y * 64;

  v8f acc[2][2];
  #pragma unroll
  for (int i = 0; i < 2; ++i)
    #pragma unroll
    for (int j = 0; j < 2; ++j)
      #pragma unroll
      for (int e = 0; e < 8; ++e) acc[i][j][e] = 0.f;

  const int ldRow = tid >> 3;   // 0..31
  const int ldC4  = tid & 7;    // float4 slot along K

#ifdef USE_ASYNC_LDS
  __shared__ float lA[2][128 * 36];   // 2 x 18.0 KB
  __shared__ float lB[2][64  * 36];   // 2 x  9.0 KB

  auto stage = [&](int b, int kk) {
    #pragma unroll
    for (int t = 0; t < 4; ++t) {
      int row = ldRow + t * 32;
      int gr  = blockRow + row;
      if (gr < M)
        async_b128(A + (size_t)gr * K + kk + ldC4 * 4, &lA[b][row * 36 + ldC4 * 4]);
    }
    #pragma unroll
    for (int t = 0; t < 2; ++t) {
      int col = ldRow + t * 32;
      async_b128(W + (size_t)(blockCol + col) * K + kk + ldC4 * 4,
                 &lB[b][col * 36 + ldC4 * 4]);
    }
  };

  stage(0, 0);   // prologue: tile 0 in flight
  int it = 0;
  for (int k0 = 0; k0 < K; k0 += 32, ++it) {
    const int cur = it & 1;
    const bool hasNext = (k0 + 32 < K);
    if (hasNext) stage(1 - cur, k0 + 32);   // overlap next DMA with this compute
    if (hasNext) wait_async_le6(); else wait_async_le0();
    __syncthreads();

    // L2 warm-up two tiles ahead (global_prefetch_b8)
    if (k0 + 64 < K) {
      int gr = blockRow + ldRow;
      if (gr < M) __builtin_prefetch(A + (size_t)gr * K + k0 + 64 + ldC4 * 4, 0, 1);
      __builtin_prefetch(W + (size_t)(blockCol + ldRow) * K + k0 + 64 + ldC4 * 4, 0, 1);
    }

    const float* bA = lA[cur];
    const float* bB = lB[cur];
#else
  __shared__ float lAs[128 * 36];
  __shared__ float lBs[64 * 36];
  for (int k0 = 0; k0 < K; k0 += 32) {
    #pragma unroll
    for (int t = 0; t < 4; ++t) {
      int row = ldRow + t * 32;
      int gr  = blockRow + row;
      float4 v = make_float4(0.f, 0.f, 0.f, 0.f);
      if (gr < M) v = *(const float4*)(A + (size_t)gr * K + k0 + ldC4 * 4);
      *(float4*)&lAs[row * 36 + ldC4 * 4] = v;
    }
    #pragma unroll
    for (int t = 0; t < 2; ++t) {
      int col = ldRow + t * 32;
      float4 v = *(const float4*)(W + (size_t)(blockCol + col) * K + k0 + ldC4 * 4);
      *(float4*)&lBs[col * 36 + ldC4 * 4] = v;
    }
    __syncthreads();
    const float* bA = lAs;
    const float* bB = lBs;
#endif

    // ---- Build fragments (ISA 7.12.2 layouts), f32 -> bf16 here ----
    union Frag { unsigned u[8]; v16bf v; };
    Frag fa[2], fb[2];
    #pragma unroll
    for (int i = 0; i < 2; ++i) {
      int row = wm * 32 + i * 16 + lrow;
      const float* s = &bA[row * 36 + lhi * 8];
      float4 x0 = *(const float4*)(s + 0);
      float4 x1 = *(const float4*)(s + 4);
      float4 x2 = *(const float4*)(s + 16);
      float4 x3 = *(const float4*)(s + 20);
      fa[i].u[0] = pkbf(x0.x, x0.y); fa[i].u[1] = pkbf(x0.z, x0.w);
      fa[i].u[2] = pkbf(x1.x, x1.y); fa[i].u[3] = pkbf(x1.z, x1.w);
      fa[i].u[4] = pkbf(x2.x, x2.y); fa[i].u[5] = pkbf(x2.z, x2.w);
      fa[i].u[6] = pkbf(x3.x, x3.y); fa[i].u[7] = pkbf(x3.z, x3.w);
    }
    #pragma unroll
    for (int j = 0; j < 2; ++j) {
      int col = wn * 32 + j * 16 + lrow;
      const float* s = &bB[col * 36 + lhi * 8];
      float4 x0 = *(const float4*)(s + 0);
      float4 x1 = *(const float4*)(s + 4);
      float4 x2 = *(const float4*)(s + 16);
      float4 x3 = *(const float4*)(s + 20);
      fb[j].u[0] = pkbf(x0.x, x0.y); fb[j].u[1] = pkbf(x0.z, x0.w);
      fb[j].u[2] = pkbf(x1.x, x1.y); fb[j].u[3] = pkbf(x1.z, x1.w);
      fb[j].u[4] = pkbf(x2.x, x2.y); fb[j].u[5] = pkbf(x2.z, x2.w);
      fb[j].u[6] = pkbf(x3.x, x3.y); fb[j].u[7] = pkbf(x3.z, x3.w);
    }

    #pragma unroll
    for (int i = 0; i < 2; ++i)
      #pragma unroll
      for (int j = 0; j < 2; ++j)
        acc[i][j] = __builtin_amdgcn_wmma_f32_16x16x32_bf16(
            false, fa[i].v, false, fb[j].v, (short)0, acc[i][j], false, false);
    __syncthreads();
  }

  // Epilogue: C/D layout = VGPR e -> rows e (lanes 0-15) / e+8 (lanes 16-31)
  #pragma unroll
  for (int i = 0; i < 2; ++i) {
    #pragma unroll
    for (int j = 0; j < 2; ++j) {
      int rBase = blockRow + wm * 32 + i * 16 + lhi * 8;
      int col   = blockCol + wn * 32 + j * 16 + lrow;
      float bv = bias ? bias[col] : 0.f;
      #pragma unroll
      for (int e = 0; e < 8; ++e) {
        int r = rBase + e;
        if (r < M) Cout[(size_t)r * ldc + col] = acc[i][j][e] + bv;
      }
    }
  }
}

// ---------------------------------------------------------------------------
// Small / sparse kernels
// ---------------------------------------------------------------------------
__global__ void k_fill_u32(unsigned* p, unsigned v, int n) {
  int i = blockIdx.x * blockDim.x + threadIdx.x;
  if (i < n) p[i] = v;
}

__global__ void k_deg(const int* __restrict__ ei, float* deg, int E) {
  int p = blockIdx.x * blockDim.x + threadIdx.x;
  if (p < E) atomicAdd(&deg[ei[p]], 1.0f);
}

// S[ei[p], :] += X[vi[p], :]   (one block per incidence pair, D/4 lanes)
__global__ void k_seg_gather_add(const float* __restrict__ X,
                                 const int* __restrict__ vi,
                                 const int* __restrict__ ei,
                                 float* __restrict__ S, int D) {
  int p = blockIdx.x;
  int v = vi[p], e = ei[p];
  int c = threadIdx.x * 4;
  float4 x = *(const float4*)(X + (size_t)v * D + c);
  float* d = S + (size_t)e * D + c;
  atomicAdd(d + 0, x.x); atomicAdd(d + 1, x.y);
  atomicAdd(d + 2, x.z); atomicAdd(d + 3, x.w);
}

__global__ void k_divdeg(float* __restrict__ S, const float* __restrict__ deg,
                         int M, int D) {
  size_t i = (size_t)blockIdx.x * blockDim.x + threadIdx.x;  // float4 index
  size_t total = (size_t)M * D / 4;
  if (i >= total) return;
  int m = (int)((i * 4) / D);
  float r = 1.0f / fmaxf(deg[m], 1.0f);
  float4* p = (float4*)S + i;
  float4 v = *p;
  v.x *= r; v.y *= r; v.z *= r; v.w *= r;
  *p = v;
}

// out[row] = dot(Y[row,:], w) (+ bias[row])   (one wave per row)
__global__ __launch_bounds__(256) void k_rowdot(
    const float* __restrict__ Y, const float* __restrict__ w,
    float* __restrict__ out, int rows, int D, const float* __restrict__ bias) {
  int row  = blockIdx.x * 8 + (threadIdx.x >> 5);
  int lane = threadIdx.x & 31;
  if (row >= rows) return;
  const float* y = Y + (size_t)row * D;
  float acc = 0.f;
  for (int c = lane * 4; c < D; c += 128) {
    float4 a = *(const float4*)(y + c);
    float4 b = *(const float4*)(w + c);
    acc += a.x * b.x + a.y * b.y + a.z * b.z + a.w * b.w;
  }
  acc = waveReduceSum(acc);
  if (!lane) out[row] = acc + (bias ? bias[row] : 0.f);
}

__global__ void k_smax_p1(const float* __restrict__ alpha,
                          const int* __restrict__ vi, const int* __restrict__ ei,
                          unsigned* __restrict__ mx, int E) {
  int p = blockIdx.x * blockDim.x + threadIdx.x;
  if (p >= E) return;
  float s = leaky02(alpha[ei[p]]);
  atomicMax(&mx[vi[p]], f32_ord(s));
}

__global__ void k_smax_p2(const float* __restrict__ alpha,
                          const int* __restrict__ vi, const int* __restrict__ ei,
                          const unsigned* __restrict__ mx,
                          float* __restrict__ sm, int E) {
  int p = blockIdx.x * blockDim.x + threadIdx.x;
  if (p >= E) return;
  float s = leaky02(alpha[ei[p]]);
  atomicAdd(&sm[vi[p]], __expf(s - ord_f32(mx[vi[p]])));
}

// Xo[vi[p], :] += w_p * Y[ei[p], :]
__global__ void k_e2v_scatter(const float* __restrict__ Y,
                              const float* __restrict__ alpha,
                              const unsigned* __restrict__ mx,
                              const float* __restrict__ sm,
                              const int* __restrict__ vi,
                              const int* __restrict__ ei,
                              float* __restrict__ Xo, int D) {
  int p = blockIdx.x;
  int v = vi[p], e = ei[p];
  float s = leaky02(alpha[e]);
  float w = __expf(s - ord_f32(mx[v])) / (sm[v] + 1e-12f);
  int c = threadIdx.x * 4;
  float4 y = *(const float4*)(Y + (size_t)e * D + c);
  float* d = Xo + (size_t)v * D + c;
  atomicAdd(d + 0, w * y.x); atomicAdd(d + 1, w * y.y);
  atomicAdd(d + 2, w * y.z); atomicAdd(d + 3, w * y.w);
}

__global__ void k_elu(float* p, size_t n) {
  size_t i = (size_t)blockIdx.x * blockDim.x + threadIdx.x;
  if (i >= n) return;
  float x = p[i];
  p[i] = x > 0.f ? x : (__expf(x) - 1.f);
}

// logits[row] = sum_j tanh(AB[row,j]) * sigmoid(AB[row,256+j]) * Wc[j] + bc
__global__ __launch_bounds__(256) void k_gated_logits(
    const float* __restrict__ AB, const float* __restrict__ Wc,
    const float* __restrict__ bc, float* __restrict__ out, int n) {
  int row  = blockIdx.x * 8 + (threadIdx.x >> 5);
  int lane = threadIdx.x & 31;
  if (row >= n) return;
  const float* a = AB + (size_t)row * 512;
  float acc = 0.f;
  for (int j = lane; j < 256; j += 32) {
    float t = tanhf(a[j]);
    float s = 1.f / (1.f + __expf(-a[256 + j]));
    acc += t * s * Wc[j];
  }
  acc = waveReduceSum(acc);
  if (!lane) out[row] = acc + bc[0];
}

__global__ void k_rmax(const float* __restrict__ l, int n, unsigned* out) {
  int i = blockIdx.x * blockDim.x + threadIdx.x;
  if (i < n) atomicMax(out, f32_ord(l[i]));
}
__global__ void k_rsumexp(const float* __restrict__ l, int n,
                          const unsigned* __restrict__ mx, float* out) {
  int i = blockIdx.x * blockDim.x + threadIdx.x;
  if (i < n) atomicAdd(out, __expf(l[i] - ord_f32(mx[0])));
}

// g[c] += softmax(l)[r] * X[r, c]   (32 rows per block, one atomic per cell)
__global__ __launch_bounds__(256) void k_wcolsum(
    const float* __restrict__ X, const float* __restrict__ l,
    const unsigned* __restrict__ mx, const float* __restrict__ sm,
    float* __restrict__ g, int n) {
  int c  = threadIdx.x * 4;               // C == 1024
  int r0 = blockIdx.x * 32;
  int r1 = min(r0 + 32, n);
  float m   = ord_f32(mx[0]);
  float inv = 1.f / sm[0];
  float4 acc = make_float4(0.f, 0.f, 0.f, 0.f);
  for (int r = r0; r < r1; ++r) {
    float w = __expf(l[r] - m) * inv;
    float4 x = *(const float4*)(X + (size_t)r * 1024 + c);
    acc.x += w * x.x; acc.y += w * x.y; acc.z += w * x.z; acc.w += w * x.w;
  }
  atomicAdd(&g[c + 0], acc.x); atomicAdd(&g[c + 1], acc.y);
  atomicAdd(&g[c + 2], acc.z); atomicAdd(&g[c + 3], acc.w);
}

__global__ __launch_bounds__(256) void k_ln1024(
    const float* __restrict__ x, const float* __restrict__ g,
    const float* __restrict__ b, float* __restrict__ out) {
  __shared__ float sh[8];
  float s = 0.f, q = 0.f;
  for (int i = threadIdx.x; i < 1024; i += 256) { float v = x[i]; s += v; q += v * v; }
  s = blockReduceSum(s, sh);
  q = blockReduceSum(q, sh);
  float mu  = s * (1.f / 1024.f);
  float var = q * (1.f / 1024.f) - mu * mu;
  float inv = rsqrtf(var + 1e-5f);
  for (int i = threadIdx.x; i < 1024; i += 256)
    out[i] = (x[i] - mu) * inv * g[i] + b[i];
}

__global__ __launch_bounds__(256) void k_final(
    const float* __restrict__ cat, const float* __restrict__ g2,
    const float* __restrict__ b2, const float* __restrict__ Wf,
    const float* __restrict__ bf, float* __restrict__ out) {
  __shared__ float sh[8];
  const int D = 6144;
  float s = 0.f, q = 0.f;
  for (int i = threadIdx.x; i < D; i += 256) { float v = cat[i]; s += v; q += v * v; }
  s = blockReduceSum(s, sh);
  q = blockReduceSum(q, sh);
  float mu  = s / (float)D;
  float var = q / (float)D - mu * mu;
  float inv = rsqrtf(var + 1e-5f);
  for (int o = 0; o < 10; ++o) {
    float acc = 0.f;
    const float* wr = Wf + (size_t)o * D;
    for (int k = threadIdx.x; k < D; k += 256) {
      float ln = (cat[k] - mu) * inv * g2[k] + b2[k];
      acc += ln * wr[k];
    }
    acc = blockReduceSum(acc, sh);
    if (threadIdx.x == 0) out[o] = acc + bf[o];
  }
}

// ---------------------------------------------------------------------------
// Host orchestration
// ---------------------------------------------------------------------------
extern "C" void kernel_launch(void* const* d_in, const int* in_sizes, int n_in,
                              void* d_out, int out_size, void* d_ws, size_t ws_size,
                              hipStream_t stream) {
  (void)n_in; (void)out_size; (void)ws_size;
  const int dC = 1024, dH = 512, nM = 5000;
  const int nV = in_sizes[0] / dC;       // 20000
  const int nE = in_sizes[1];            // 160000

  const float* Xb[3] = {(const float*)d_in[0], (const float*)d_in[3], (const float*)d_in[6]};
  const int*   VI[3] = {(const int*)d_in[1], (const int*)d_in[4], (const int*)d_in[7]};
  const int*   EI[3] = {(const int*)d_in[2], (const int*)d_in[5], (const int*)d_in[8]};
  const float* Wt0 = (const float*)d_in[9];   const float* bt0 = (const float*)d_in[10];
  const float* we0 = (const float*)d_in[11];
  const float* Wt1 = (const float*)d_in[12];  const float* bt1 = (const float*)d_in[13];
  const float* we1 = (const float*)d_in[14];
  const float* Wa  = (const float*)d_in[15];  const float* ba  = (const float*)d_in[16];
  const float* Wb  = (const float*)d_in[17];  const float* bb  = (const float*)d_in[18];
  const float* Wc  = (const float*)d_in[19];  const float* bc  = (const float*)d_in[20];
  const float* Wout= (const float*)d_in[21];  const float* bout= (const float*)d_in[22];
  const float* g_bn = (const float*)d_in[23]; const float* b_bn = (const float*)d_in[24];
  const float* g_bn2= (const float*)d_in[25]; const float* b_bn2= (const float*)d_in[26];
  const float* Wf  = (const float*)d_in[27];  const float* bf  = (const float*)d_in[28];

  // Workspace layout (floats)
  float* ws = (float*)d_ws;
  size_t o = 0;
  float* R1 = ws + o; o += (size_t)nV * dC;   // T1 / h
  float* R2 = ws + o; o += (size_t)nM * dC;   // Y1 / y
  float* R3 = ws + o; o += (size_t)nV * dH;   // T0 / h1 / AB
  float* R4 = ws + o; o += (size_t)nM * dH;   // Y0
  float* deg    = ws + o; o += nM;
  float* alpha  = ws + o; o += nM;
  unsigned* maxv = (unsigned*)(ws + o); o += nV;
  float* sumv   = ws + o; o += nV;
  float* logits = ws + o; o += nV;
  unsigned* scalU = (unsigned*)(ws + o); o += 4;
  float* scalF  = ws + o; o += 4;
  float* g1     = ws + o; o += 1024;
  float* gv     = ws + o; o += 1024;
  float* cat    = ws + o; o += 8192;

  auto run_attn = [&](const float* x, int n, float* catSlot) {
    // AB = [tanh-gate | sigmoid-gate] pre-activations, stacked into R3 [n,512]
    k_gemm_bf16<<<dim3((n + 127) / 128, 4), 256, 0, stream>>>(x, Wa, ba, R3,       n, 256, dC, 512);
    k_gemm_bf16<<<dim3((n + 127) / 128, 4), 256, 0, stream>>>(x, Wb, bb, R3 + 256, n, 256, dC, 512);
    k_gated_logits<<<(n + 7) / 8, 256, 0, stream>>>(R3, Wc, bc, logits, n);
    k_fill_u32<<<1, 32, 0, stream>>>(scalU, ORD_NEG_INF, 1);
    (void)hipMemsetAsync(scalF, 0, sizeof(float), stream);
    k_rmax<<<(n + 255) / 256, 256, 0, stream>>>(logits, n, scalU);
    k_rsumexp<<<(n + 255) / 256, 256, 0, stream>>>(logits, n, scalU, scalF);
    (void)hipMemsetAsync(g1, 0, 1024 * sizeof(float), stream);
    k_wcolsum<<<(n + 31) / 32, 256, 0, stream>>>(x, logits, scalU, scalF, g1, n);
    k_rowdot<<<1024 / 8, 256, 0, stream>>>(Wout, g1, gv, 1024, 1024, bout);
    k_ln1024<<<1, 256, 0, stream>>>(gv, g_bn, b_bn, catSlot);
  };

  for (int br = 0; br < 3; ++br) {
    const float* X = Xb[br]; const int* vi = VI[br]; const int* ei = EI[br];

    // ---- UniGAT layer 1 (C=1024 -> H=512) ----
    k_gemm_bf16<<<dim3((nV + 127) / 128, dH / 64), 256, 0, stream>>>(X, Wt0, bt0, R3, nV, dH, dC, dH);
    (void)hipMemsetAsync(R4, 0, (size_t)nM * dH * sizeof(float), stream);
    (void)hipMemsetAsync(deg, 0, nM * sizeof(float), stream);
    k_deg<<<(nE + 255) / 256, 256, 0, stream>>>(ei, deg, nE);
    k_seg_gather_add<<<nE, dH / 4, 0, stream>>>(R3, vi, ei, R4, dH);
    k_divdeg<<<(unsigned)(((size_t)nM * dH / 4 + 255) / 256), 256, 0, stream>>>(R4, deg, nM, dH);
    k_rowdot<<<(nM + 7) / 8, 256, 0, stream>>>(R4, we0, alpha, nM, dH, nullptr);
    k_fill_u32<<<(nV + 255) / 256, 256, 0, stream>>>(maxv, ORD_NEG_INF, nV);
    (void)hipMemsetAsync(sumv, 0, nV * sizeof(float), stream);
    k_smax_p1<<<(nE + 255) / 256, 256, 0, stream>>>(alpha, vi, ei, maxv, nE);
    k_smax_p2<<<(nE + 255) / 256, 256, 0, stream>>>(alpha, vi, ei, maxv, sumv, nE);
    (void)hipMemsetAsync(R3, 0, (size_t)nV * dH * sizeof(float), stream);
    k_e2v_scatter<<<nE, dH / 4, 0, stream>>>(R4, alpha, maxv, sumv, vi, ei, R3, dH);
    k_elu<<<(unsigned)(((size_t)nV * dH + 255) / 256), 256, 0, stream>>>(R3, (size_t)nV * dH);

    // ---- UniGAT layer 2 (H=512 -> C=1024), no elu ----
    k_gemm_bf16<<<dim3((nV + 127) / 128, dC / 64), 256, 0, stream>>>(R3, Wt1, bt1, R1, nV, dC, dH, dC);
    (void)hipMemsetAsync(R2, 0, (size_t)nM * dC * sizeof(float), stream);
    k_seg_gather_add<<<nE, dC / 4, 0, stream>>>(R1, vi, ei, R2, dC);
    k_divdeg<<<(unsigned)(((size_t)nM * dC / 4 + 255) / 256), 256, 0, stream>>>(R2, deg, nM, dC);
    k_rowdot<<<(nM + 7) / 8, 256, 0, stream>>>(R2, we1, alpha, nM, dC, nullptr);
    k_fill_u32<<<(nV + 255) / 256, 256, 0, stream>>>(maxv, ORD_NEG_INF, nV);
    (void)hipMemsetAsync(sumv, 0, nV * sizeof(float), stream);
    k_smax_p1<<<(nE + 255) / 256, 256, 0, stream>>>(alpha, vi, ei, maxv, nE);
    k_smax_p2<<<(nE + 255) / 256, 256, 0, stream>>>(alpha, vi, ei, maxv, sumv, nE);
    (void)hipMemsetAsync(R1, 0, (size_t)nV * dC * sizeof(float), stream);
    k_e2v_scatter<<<nE, dC / 4, 0, stream>>>(R2, alpha, maxv, sumv, vi, ei, R1, dC);

    // ---- y = v2e_mean(h) ----
    (void)hipMemsetAsync(R2, 0, (size_t)nM * dC * sizeof(float), stream);
    k_seg_gather_add<<<nE, dC / 4, 0, stream>>>(R1, vi, ei, R2, dC);
    k_divdeg<<<(unsigned)(((size_t)nM * dC / 4 + 255) / 256), 256, 0, stream>>>(R2, deg, nM, dC);

    // ---- gated attention pooling + LN ----
    run_attn(R1, nV, cat + br * 1024);        // gx -> xs[br]
    run_attn(R2, nM, cat + (3 + br) * 1024);  // gy -> ys[br]
  }

  k_final<<<1, 256, 0, stream>>>(cat, g_bn2, b_bn2, Wf, bf, (float*)d_out);
}
